// PlasmidLMAttention_17257178595377
// MI455X (gfx1250) — matrix operand order repaired and checked by reading, not compiled, not run on hardware
//
#include <hip/hip_runtime.h>
#include <hip/hip_bf16.h>

// ---------------------------------------------------------------------------
// PlasmidLMAttention for MI455X (gfx1250, wave32, WMMA + TDM)
// Q/K/V/O projections as TDM-staged LDS block-tiled bf16 WMMA GEMMs,
// causal flash attention on v_wmma_f32_16x16x32_bf16.
// ---------------------------------------------------------------------------

typedef __bf16 bf16;
typedef __attribute__((ext_vector_type(16))) __bf16 v16bf;
typedef __attribute__((ext_vector_type(8)))  float  v8f;
typedef __attribute__((ext_vector_type(4)))  unsigned int v4u;
typedef __attribute__((ext_vector_type(8)))  int    v8i;

#define HID   2048
#define NH    16
#define HD    128
#define BB    4
#define SS    2048
#define MROWS (BB * SS)   // 8192

// ---- WMMA fragment helpers (layouts per CDNA5 ISA 7.12.2, wave32) ----------

// A-matrix 16x32 bf16: lane l holds row m=l%16; lanes 0-15 K=0..7,16..23,
// lanes 16-31 K=8..15,24..31 (2 elems per VGPR).
__device__ inline v16bf load_frag_a(const bf16* base, int stride) {
  int l = threadIdx.x & 31;
  int m = l & 15;
  int kb = (l >> 4) << 3;              // 0 or 8
  const bf16* row = base + (size_t)m * stride + kb;
  v16bf a;
#pragma unroll
  for (int vg = 0; vg < 8; ++vg) {
    int k = (vg < 4) ? (2 * vg) : (16 + 2 * (vg - 4));
    a[2 * vg]     = row[k];
    a[2 * vg + 1] = row[k + 1];
  }
  return a;
}

// B-matrix 32x16 bf16 where B[k][n] = src[n*stride + k] (src rows are N):
// lane l holds col n=l%16; lanes 0-15 K=0..7,16..23, lanes 16-31 K=8..15,24..31.
__device__ inline v16bf load_frag_bt(const bf16* base, int stride) {
  int l = threadIdx.x & 31;
  int n = l & 15;
  int kb = (l >> 4) << 3;              // 0 or 8
  const bf16* p = base + (size_t)n * stride + kb;
  v16bf b;
#pragma unroll
  for (int i = 0; i < 16; ++i) b[i] = p[i + ((i >> 3) << 3)];
  return b;
}

__device__ inline v8f wmma_bf16(v16bf a, v16bf b, v8f c) {
  return __builtin_amdgcn_wmma_f32_16x16x32_bf16(
      false, a, false, b, (short)0, c, false, false);
}

// ---- Tensor Data Mover: 2D tile (tile_rows x 32 bf16) global -> LDS --------
// D# per CDNA5 ISA 8.3/8.4. 2-group form (VADDR2/3 = NULL -> up to 2D).
// Must be issued with wave-uniform operands (compiler keeps them in SGPRs).

__device__ __attribute__((always_inline)) inline void
tdm_load_tile_2d(unsigned lds_addr, const bf16* gptr, unsigned tile_rows,
                 unsigned row_stride_elems) {
  unsigned long long ga = (unsigned long long)(uintptr_t)gptr;
  v4u g0;
  g0[0] = 1u;                                        // count=1, user mode
  g0[1] = lds_addr;                                  // lds_addr [63:32]
  g0[2] = (unsigned)ga;                              // global_addr lo
  g0[3] = (unsigned)(ga >> 32) | 0x80000000u;        // addr hi | type=2
  v8i g1;
  g1[0] = 0x10000;                                   // data_size=1 (2 bytes)
  g1[1] = (int)((row_stride_elems & 0xFFFFu) << 16); // tensor_dim0[15:0]
  g1[2] = (int)((row_stride_elems >> 16) & 0xFFFFu); // tensor_dim0[31:16]
  g1[3] = (int)(16u | (32u << 16));                  // tensor_dim1=1M, tile_dim0=32
  g1[4] = (int)tile_rows;                            // tile_dim1 (tile_dim2=0)
  g1[5] = (int)row_stride_elems;                     // tensor_dim0_stride lo
  g1[6] = 0;                                         // stride0 hi / stride1 lo
  g1[7] = 0;                                         // stride1 hi
  asm volatile("tensor_load_to_lds %0, %1" : : "s"(g0), "s"(g1) : "memory");
}

// ---- elementwise f32 -> bf16 ----------------------------------------------

__global__ void cvt_f32_bf16(const float* __restrict__ s, bf16* __restrict__ d,
                             int n) {
  int i = blockIdx.x * blockDim.x + threadIdx.x;
  if (i < n) d[i] = (bf16)s[i];
}

// ---- GEMM: C[m][n] = sum_k A[m][k] * W[n][k]  (A,W bf16; C f32) ------------
// Block tile 128x64, BK=32, double-buffered LDS fed by TDM (wave 0 issues),
// 8 waves x (16 rows x 64 cols) compute from LDS.

__global__ void __launch_bounds__(256)
gemm_bf16_f32(const bf16* __restrict__ A, const bf16* __restrict__ W,
              float* __restrict__ C, int M, int N, int K) {
  __shared__ bf16 As[2][128 * 32];   // 2 x 8KB
  __shared__ bf16 Bs[2][64 * 32];    // 2 x 4KB

  int ntb = N >> 6;
  int nb  = (blockIdx.x % ntb) << 6;
  int mb  = (blockIdx.x / ntb) * 128;
  if (mb >= M) return;

  int wv = threadIdx.x >> 5;         // wave 0..7 -> 16-row strip
  const bf16* Abase = A + (size_t)mb * K;
  const bf16* Wbase = W + (size_t)nb * K;

  unsigned ldsA[2] = {(unsigned)(uintptr_t)&As[0][0],
                      (unsigned)(uintptr_t)&As[1][0]};
  unsigned ldsB[2] = {(unsigned)(uintptr_t)&Bs[0][0],
                      (unsigned)(uintptr_t)&Bs[1][0]};

  if (threadIdx.x < 32) {            // wave 0 drives the TDM
    tdm_load_tile_2d(ldsA[0], Abase, 128, (unsigned)K);
    tdm_load_tile_2d(ldsB[0], Wbase, 64, (unsigned)K);
  }

  v8f acc0 = {}, acc1 = {}, acc2 = {}, acc3 = {};
  int buf = 0;

  for (int k0 = 0; k0 < K; k0 += 32) {
    if (threadIdx.x < 32) {
      if (k0 + 32 < K) {
        // prefetch next BK slab into the other buffer, then wait for current
        tdm_load_tile_2d(ldsA[buf ^ 1], Abase + k0 + 32, 128, (unsigned)K);
        tdm_load_tile_2d(ldsB[buf ^ 1], Wbase + k0 + 32, 64, (unsigned)K);
        __builtin_amdgcn_s_wait_tensorcnt(2);   // current buffer's 2 ops done
      } else {
        __builtin_amdgcn_s_wait_tensorcnt(0);
      }
    }
    __syncthreads();                 // publish TDM-written LDS to all waves

    v16bf a  = load_frag_a(&As[buf][wv * 16 * 32], 32);
    v16bf b0 = load_frag_bt(&Bs[buf][0 * 16 * 32], 32);
    v16bf b1 = load_frag_bt(&Bs[buf][1 * 16 * 32], 32);
    v16bf b2 = load_frag_bt(&Bs[buf][2 * 16 * 32], 32);
    v16bf b3 = load_frag_bt(&Bs[buf][3 * 16 * 32], 32);
    acc0 = wmma_bf16(a, b0, acc0);
    acc1 = wmma_bf16(a, b1, acc1);
    acc2 = wmma_bf16(a, b2, acc2);
    acc3 = wmma_bf16(a, b3, acc3);

    __syncthreads();                 // all reads done before buffer reuse
    buf ^= 1;
  }

  // C/D layout: VGPR r -> M = r (+8 for lanes 16-31), N = lane%16.
  int l  = threadIdx.x & 31;
  int n  = l & 15;
  int h2 = l >> 4;
#pragma unroll
  for (int r = 0; r < 8; ++r) {
    int m = mb + wv * 16 + r + 8 * h2;
    float* crow = C + (size_t)m * N + nb + n;
    crow[0]  = acc0[r];
    crow[16] = acc1[r];
    crow[32] = acc2[r];
    crow[48] = acc3[r];
  }
}

// ---- RoPE + pack: fp32 K,V outputs; bf16 Q(scaled)/K (b,h,s,d); ------------
// bf16 V transposed to (b,h,d,s) so PV B-fragments load contiguously.

__global__ void rope_pack(const float* __restrict__ Qf,
                          const float* __restrict__ Kf,
                          const float* __restrict__ Vf,
                          const float* __restrict__ cosb,
                          const float* __restrict__ sinb,
                          const int*   __restrict__ offp,
                          float* __restrict__ kout, float* __restrict__ vout,
                          bf16* __restrict__ qbf, bf16* __restrict__ kbf,
                          bf16* __restrict__ vtbf) {
  int idx = blockIdx.x * blockDim.x + threadIdx.x;   // B*H*S*(HD/2)
  int j = idx & 63;
  int t = idx >> 6;
  int s = t & (SS - 1); t >>= 11;
  int h = t & (NH - 1);
  int b = t >> 4;
  int off = offp[0];

  float c  = cosb[(size_t)(s + off) * (HD / 2) + j];
  float sn = sinb[(size_t)(s + off) * (HD / 2) + j];

  size_t rc = ((size_t)b * SS + s) * HID + h * HD + 2 * j;
  float q1 = Qf[rc], q2 = Qf[rc + 1];
  float k1 = Kf[rc], k2 = Kf[rc + 1];
  float v1 = Vf[rc], v2 = Vf[rc + 1];

  float qr1 = q1 * c - q2 * sn, qr2 = q1 * sn + q2 * c;
  float kr1 = k1 * c - k2 * sn, kr2 = k1 * sn + k2 * c;

  size_t hd = (((size_t)b * NH + h) * SS + s) * HD + 2 * j;
  kout[hd] = kr1; kout[hd + 1] = kr2;
  vout[hd] = v1;  vout[hd + 1] = v2;

  const float qs = 0.08838834764831845f;  // 1/sqrt(HD), folded into Q
  qbf[hd]     = (bf16)(qr1 * qs);
  qbf[hd + 1] = (bf16)(qr2 * qs);
  kbf[hd]     = (bf16)kr1;
  kbf[hd + 1] = (bf16)kr2;

  size_t vt = (((size_t)b * NH + h) * HD + 2 * j) * SS + s;
  vtbf[vt]      = (bf16)v1;
  vtbf[vt + SS] = (bf16)v2;
}

// ---- causal flash attention: one 32-thread wave per (b,h,16 q rows) --------

__global__ void __launch_bounds__(32)
attn_kernel(const bf16* __restrict__ Qbf, const bf16* __restrict__ Kbf,
            const bf16* __restrict__ Vtbf, bf16* __restrict__ attn) {
  __shared__ bf16 lp[16 * 32];         // P tile staging (C-layout -> A-layout)

  int qt = blockIdx.x & (SS / 16 - 1); // 128 q-tiles
  int bh = blockIdx.x >> 7;
  int h  = bh & (NH - 1);
  int b  = bh >> 4;
  int qbase = qt * 16;

  const bf16* Qb = Qbf  + ((size_t)bh * SS + qbase) * HD;
  const bf16* Kb = Kbf  + (size_t)bh * SS * HD;
  const bf16* Vt = Vtbf + (size_t)bh * HD * SS;

  v16bf aq[4];
#pragma unroll
  for (int c = 0; c < 4; ++c) aq[c] = load_frag_a(Qb + 32 * c, HD);

  v8f o[8];
  float rowM[8], rowL[8];
#pragma unroll
  for (int f = 0; f < 8; ++f) o[f] = v8f{};
#pragma unroll
  for (int r = 0; r < 8; ++r) { rowM[r] = -1e30f; rowL[r] = 0.0f; }

  int l  = threadIdx.x & 31;
  int n  = l & 15;
  int h2 = l >> 4;

  for (int tb = 0; tb <= qbase + 15; tb += 32) {
    const bf16* K0 = Kb + (size_t)tb * HD;

    // scores: S0 = t in [tb,tb+16), S1 = t in [tb+16,tb+32)
    v8f s0 = {}, s1 = {};
#pragma unroll
    for (int c = 0; c < 4; ++c) {
      s0 = wmma_bf16(aq[c], load_frag_bt(K0 + c * 32, HD), s0);
      s1 = wmma_bf16(aq[c], load_frag_bt(K0 + (size_t)16 * HD + c * 32, HD), s1);
    }

    // online softmax (scale already folded into Q)
    float p0[8], p1[8];
#pragma unroll
    for (int r = 0; r < 8; ++r) {
      int q = qbase + r + 8 * h2;
      float x0 = (tb + n      <= q) ? s0[r] : -1e30f;
      float x1 = (tb + 16 + n <= q) ? s1[r] : -1e30f;
      float mx = fmaxf(x0, x1);
#pragma unroll
      for (int d = 1; d < 16; d <<= 1) mx = fmaxf(mx, __shfl_xor(mx, d, 32));
      float mn = fmaxf(rowM[r], mx);
      float al = __expf(rowM[r] - mn);
      float e0 = __expf(x0 - mn);
      float e1 = __expf(x1 - mn);
      float rs = e0 + e1;
#pragma unroll
      for (int d = 1; d < 16; d <<= 1) rs += __shfl_xor(rs, d, 32);
      rowL[r] = rowL[r] * al + rs;
      rowM[r] = mn;
      p0[r] = e0; p1[r] = e1;
#pragma unroll
      for (int f = 0; f < 8; ++f) o[f][r] *= al;
    }

    // restage P (C layout) into LDS as a row-major 16x32 tile
    __syncthreads();
#pragma unroll
    for (int r = 0; r < 8; ++r) {
      lp[(r + 8 * h2) * 32 + n]      = (bf16)p0[r];
      lp[(r + 8 * h2) * 32 + 16 + n] = (bf16)p1[r];
    }
    __syncthreads();

    v16bf ap = load_frag_a(lp, 32);
#pragma unroll
    for (int f = 0; f < 8; ++f) {
      v16bf bv = load_frag_bt(Vt + (size_t)(f * 16) * SS + tb, SS);
      o[f] = wmma_bf16(ap, bv, o[f]);
    }
  }

  // epilogue: normalize, write bf16 in (b, s, h*HD+d) layout for O-projection
#pragma unroll
  for (int r = 0; r < 8; ++r) {
    float inv = 1.0f / rowL[r];
    int q = qbase + r + 8 * h2;
    size_t base = ((size_t)b * SS + q) * HID + h * HD + n;
#pragma unroll
    for (int f = 0; f < 8; ++f)
      attn[base + (size_t)f * 16] = (bf16)(o[f][r] * inv);
  }
}

// ---------------------------------------------------------------------------

extern "C" void kernel_launch(void* const* d_in, const int* in_sizes, int n_in,
                              void* d_out, int out_size, void* d_ws,
                              size_t ws_size, hipStream_t stream) {
  const float* hs  = (const float*)d_in[0];
  const float* rc  = (const float*)d_in[1];
  const float* rs  = (const float*)d_in[2];
  const float* Wq  = (const float*)d_in[3];
  const float* Wk  = (const float*)d_in[4];
  const float* Wv  = (const float*)d_in[5];
  const float* Wo  = (const float*)d_in[6];
  const int*   off = (const int*)d_in[7];

  float* out  = (float*)d_out;                         // (B,S,HID)
  float* kout = out  + (size_t)BB * SS * HID;          // (B,H,S,D)
  float* vout = kout + (size_t)BB * NH * SS * HD;      // (B,H,S,D)

  char* ws = (char*)d_ws;
  size_t o = 0;
  bf16* Xbf = (bf16*)(ws + o); o += (size_t)MROWS * HID * sizeof(bf16);
  bf16* Wqb = (bf16*)(ws + o); o += (size_t)HID * HID * sizeof(bf16);
  bf16* Wkb = (bf16*)(ws + o); o += (size_t)HID * HID * sizeof(bf16);
  bf16* Wvb = (bf16*)(ws + o); o += (size_t)HID * HID * sizeof(bf16);
  bf16* Wob = (bf16*)(ws + o); o += (size_t)HID * HID * sizeof(bf16);
  float* Qf = (float*)(ws + o); o += (size_t)MROWS * HID * sizeof(float);
  float* Kf = (float*)(ws + o); o += (size_t)MROWS * HID * sizeof(float);
  float* Vf = (float*)(ws + o); o += (size_t)MROWS * HID * sizeof(float);
  bf16* Qbf  = (bf16*)(ws + o); o += (size_t)MROWS * HID * sizeof(bf16);
  bf16* Kbf  = (bf16*)(ws + o); o += (size_t)MROWS * HID * sizeof(bf16);
  bf16* Vtbf = (bf16*)(ws + o); o += (size_t)MROWS * HID * sizeof(bf16);
  bf16* Abf  = (bf16*)Qf;      // Qf32 dead after rope_pack -> reuse

  // 1) fp32 -> bf16 conversions
  int nx = MROWS * HID;                      // 16,777,216
  cvt_f32_bf16<<<nx / 256, 256, 0, stream>>>(hs, Xbf, nx);
  int nw = HID * HID;                        // 4,194,304
  cvt_f32_bf16<<<nw / 256, 256, 0, stream>>>(Wq, Wqb, nw);
  cvt_f32_bf16<<<nw / 256, 256, 0, stream>>>(Wk, Wkb, nw);
  cvt_f32_bf16<<<nw / 256, 256, 0, stream>>>(Wv, Wvb, nw);
  cvt_f32_bf16<<<nw / 256, 256, 0, stream>>>(Wo, Wob, nw);

  // 2) Q/K/V projections: (M/128)*(N/64) = 64*32 = 2048 blocks of 8 waves
  gemm_bf16_f32<<<2048, 256, 0, stream>>>(Xbf, Wqb, Qf, MROWS, HID, HID);
  gemm_bf16_f32<<<2048, 256, 0, stream>>>(Xbf, Wkb, Kf, MROWS, HID, HID);
  gemm_bf16_f32<<<2048, 256, 0, stream>>>(Xbf, Wvb, Vf, MROWS, HID, HID);

  // 3) RoPE + fp32 K/V outputs + bf16 packing (V transposed)
  int nr = BB * NH * SS * (HD / 2);          // 8,388,608
  rope_pack<<<nr / 256, 256, 0, stream>>>(Qf, Kf, Vf, rc, rs, off, kout, vout,
                                          Qbf, Kbf, Vtbf);

  // 4) causal flash attention: one wave per (b,h,16-row q tile)
  attn_kernel<<<BB * NH * (SS / 16), 32, 0, stream>>>(Qbf, Kbf, Vtbf, Abf);

  // 5) output projection -> d_out
  gemm_bf16_f32<<<2048, 256, 0, stream>>>(Abf, Wob, out, MROWS, HID, HID);
}
